// MultiHeadAtttention_46797963657634
// MI455X (gfx1250) — matrix operand order repaired and checked by reading, not compiled, not run on hardware
//
#include <hip/hip_runtime.h>
#include <hip/hip_bf16.h>

// ---------------- problem constants ----------------
#define BATCH   4
#define SEQ     2048
#define DMODEL  1024
#define HEADS   16
#define DK      64
#define MROWS   (BATCH * SEQ)       // 8192
#define SCALE_Q 0.125f              // 1/sqrt(64)

typedef __attribute__((ext_vector_type(16))) __bf16 v16bf;
typedef __attribute__((ext_vector_type(8)))  __bf16 v8bf;
typedef __attribute__((ext_vector_type(8)))  float  v8f;
typedef __attribute__((ext_vector_type(4)))  float  v4f;

// ---------------- WMMA helper ----------------
__device__ __forceinline__ v8f wmma_bf16(v16bf a, v16bf b, v8f c) {
    // D = A(16x32 bf16) x B(32x16 bf16) + C(16x16 f32)
    return __builtin_amdgcn_wmma_f32_16x16x32_bf16(
        /*neg_a=*/false, a, /*neg_b=*/false, b,
        /*c_mod=*/(short)0, c, /*reuse_a=*/false, /*reuse_b=*/false);
}

__device__ __forceinline__ v16bf combine16(v8bf lo, v8bf hi) {
    v16bf r;
#pragma unroll
    for (int i = 0; i < 8; ++i) { r[i] = lo[i]; r[i + 8] = hi[i]; }
    return r;
}

// A fragment (16x32, 16-bit): lane<16 -> row=lane, k in [0..7]+[16..23];
// lane>=16 -> row=lane-16, k in [8..15]+[24..31].
__device__ __forceinline__ v16bf load_a_bf16(const __bf16* base, int ld, int lane) {
    int row = lane & 15, half = (lane >> 4) & 1;
    const __bf16* q = base + row * ld + 8 * half;
    v8bf lo = *(const v8bf*)(q);
    v8bf hi = *(const v8bf*)(q + 16);
    return combine16(lo, hi);
}

// Same A fragment but sourced from f32 memory with on-the-fly bf16 convert.
__device__ __forceinline__ v16bf load_a_f32(const float* base, int ld, int lane) {
    int row = lane & 15, half = (lane >> 4) & 1;
    const float* q = base + row * ld + 8 * half;
    v4f x0 = *(const v4f*)(q);
    v4f x1 = *(const v4f*)(q + 4);
    v4f y0 = *(const v4f*)(q + 16);
    v4f y1 = *(const v4f*)(q + 20);
    v16bf r;
#pragma unroll
    for (int i = 0; i < 4; ++i) {
        r[i]      = (__bf16)x0[i];
        r[i + 4]  = (__bf16)x1[i];
        r[i + 8]  = (__bf16)y0[i];
        r[i + 12] = (__bf16)y1[i];
    }
    return r;
}

// B fragment (32x16, 16-bit) from TRANSPOSED storage T[n][k] (row = output col n):
// lane n = lane&15, kh = lane>>4, k in [8kh..8kh+7]+[16+8kh..16+8kh+7] (matches A).
__device__ __forceinline__ v16bf load_b_t(const __bf16* base, int ld, int lane) {
    int n = lane & 15, kh = (lane >> 4) & 1;
    const __bf16* q = base + n * ld + 8 * kh;
    v8bf lo = *(const v8bf*)(q);
    v8bf hi = *(const v8bf*)(q + 16);
    return combine16(lo, hi);
}

// ---------------- 1) weight transpose + bf16 convert: Wt[n][k] = W[k][n] ----------------
__global__ __launch_bounds__(256) void transpose_w_kernel(const float* __restrict__ W,
                                                          __bf16* __restrict__ Wt) {
    __shared__ float tile[32][33];
    int tx = threadIdx.x & 31;
    int ty = threadIdx.x >> 5;          // 0..7
    int k0 = blockIdx.y * 32;
    int n0 = blockIdx.x * 32;
#pragma unroll
    for (int r = ty; r < 32; r += 8)
        tile[r][tx] = W[(size_t)(k0 + r) * DMODEL + (n0 + tx)];
    __syncthreads();
#pragma unroll
    for (int r = ty; r < 32; r += 8)
        Wt[(size_t)(n0 + r) * DMODEL + (k0 + tx)] = (__bf16)tile[tx][r];
}

// ---------------- 2) projection / output GEMM ----------------
// C[m,n] = (sum_k A[m,k] * W[k,n] + bias[n]) * scale
// Wave tile 64(M) x 32(N); block tile 128 x 128 (8 waves as 2 M-groups x 4 N-groups).
// OUT_MODE 0: bf16 out, layout [B,H,S,DK]          (Q, K)
// OUT_MODE 1: bf16 out, layout [B,H,DK,S] (transp) (V)
// OUT_MODE 2: f32  out, layout [M, DMODEL]         (final projection)
template <int OUT_MODE, bool A_F32>
__global__ __launch_bounds__(256) void gemm_kernel(const void* __restrict__ Ap,
                                                   const __bf16* __restrict__ Wt,
                                                   const float* __restrict__ bias,
                                                   void* __restrict__ Cp,
                                                   float scale) {
    int lane = threadIdx.x & 31;
    int wave = threadIdx.x >> 5;        // 0..7
    int wm = wave & 1;                  // 0..1 (M)
    int wn = wave >> 1;                 // 0..3 (N)
    int rb = blockIdx.y * 128 + wm * 64;
    int cb = blockIdx.x * 128 + wn * 32;

    v8f acc[4][2] = {};
    for (int k = 0; k < DMODEL; k += 32) {
        v16bf a[4];
#pragma unroll
        for (int i = 0; i < 4; ++i) {
            if (A_F32) {
                const float* A = (const float*)Ap;
                a[i] = load_a_f32(A + (size_t)(rb + 16 * i) * DMODEL + k, DMODEL, lane);
            } else {
                const __bf16* A = (const __bf16*)Ap;
                a[i] = load_a_bf16(A + (size_t)(rb + 16 * i) * DMODEL + k, DMODEL, lane);
            }
        }
        v16bf b0 = load_b_t(Wt + (size_t)cb * DMODEL + k, DMODEL, lane);
        v16bf b1 = load_b_t(Wt + (size_t)(cb + 16) * DMODEL + k, DMODEL, lane);
#pragma unroll
        for (int i = 0; i < 4; ++i) {
            acc[i][0] = wmma_bf16(a[i], b0, acc[i][0]);
            acc[i][1] = wmma_bf16(a[i], b1, acc[i][1]);
        }
    }

    // Epilogue: C layout -> lane holds column n = lane&15, rows r + 8*(lane>>4)
    int nl = lane & 15, half = (lane >> 4) & 1;
#pragma unroll
    for (int i = 0; i < 4; ++i) {
#pragma unroll
        for (int j = 0; j < 2; ++j) {
            int ncol = cb + 16 * j + nl;
            float bv = bias[ncol];
#pragma unroll
            for (int r = 0; r < 8; ++r) {
                int mrow = rb + 16 * i + r + 8 * half;
                float val = (acc[i][j][r] + bv) * scale;
                if (OUT_MODE == 2) {
                    ((float*)Cp)[(size_t)mrow * DMODEL + ncol] = val;
                } else {
                    int b = mrow >> 11, s = mrow & (SEQ - 1);
                    int h = ncol >> 6, d = ncol & (DK - 1);
                    size_t idx = (OUT_MODE == 0)
                        ? (((size_t)(b * HEADS + h) * SEQ + s) * DK + d)
                        : (((size_t)(b * HEADS + h) * DK + d) * SEQ + s);
                    ((__bf16*)Cp)[idx] = (__bf16)val;
                }
            }
        }
    }
}

// ---------------- 3) column softmax stats (softmax over QUERY axis i) ----------------
// For each (b,h,j): m[j] = max_i s[i,j], l[j] = sum_i exp(s[i,j]-m[j]).
// Each wave owns 32 j-columns (two 16-wide subtiles, B-fragments resident).
__global__ __launch_bounds__(256) void attn_stats_kernel(const __bf16* __restrict__ Q,
                                                         const __bf16* __restrict__ Kmat,
                                                         float* __restrict__ Mcol,
                                                         float* __restrict__ Lcol) {
    int bh = blockIdx.y;
    int lane = threadIdx.x & 31;
    int wave = threadIdx.x >> 5;
    int j0 = blockIdx.x * 256 + wave * 32;

    const __bf16* Qh = Q    + (size_t)bh * SEQ * DK;
    const __bf16* Kh = Kmat + (size_t)bh * SEQ * DK;

    // Resident B fragments: B[k=d][n=j] = K[j0+16*jj+n][d]
    v16bf bk[2][2];
#pragma unroll
    for (int jj = 0; jj < 2; ++jj) {
        bk[jj][0] = load_b_t(Kh + (size_t)(j0 + 16 * jj) * DK + 0,  DK, lane);
        bk[jj][1] = load_b_t(Kh + (size_t)(j0 + 16 * jj) * DK + 32, DK, lane);
    }

    float m[2] = {-1e30f, -1e30f};
    float l[2] = {0.0f, 0.0f};
    for (int i = 0; i < SEQ; i += 16) {
        v16bf a0 = load_a_bf16(Qh + (size_t)i * DK + 0,  DK, lane);
        v16bf a1 = load_a_bf16(Qh + (size_t)i * DK + 32, DK, lane);
#pragma unroll
        for (int jj = 0; jj < 2; ++jj) {
            v8f c = {};
            c = wmma_bf16(a0, bk[jj][0], c);
            c = wmma_bf16(a1, bk[jj][1], c);
            float tm = c[0];
#pragma unroll
            for (int r = 1; r < 8; ++r) tm = fmaxf(tm, c[r]);
            float nm = fmaxf(m[jj], tm);
            float sum = 0.0f;
#pragma unroll
            for (int r = 0; r < 8; ++r) sum += __expf(c[r] - nm);
            l[jj] = l[jj] * __expf(m[jj] - nm) + sum;
            m[jj] = nm;
        }
    }
    // merge the two lane-halves (lane and lane^16 hold the same column)
#pragma unroll
    for (int jj = 0; jj < 2; ++jj) {
        float mo = __shfl_xor(m[jj], 16, 32);
        float lo = __shfl_xor(l[jj], 16, 32);
        float nm = fmaxf(m[jj], mo);
        float lt = l[jj] * __expf(m[jj] - nm) + lo * __expf(mo - nm);
        if (lane < 16) {
            Mcol[(size_t)bh * SEQ + j0 + 16 * jj + lane] = nm;
            Lcol[(size_t)bh * SEQ + j0 + 16 * jj + lane] = lt;
        }
    }
}

// ---------------- 4) apply: ctx[i,:] = sum_j exp(s[i,j]-m_j)/l_j * V[j,:] ----------------
// Each wave owns 32 i-rows (two 16-row subtiles); per 32-j step: 8 score WMMAs,
// exp through a per-wave LDS tile (C-layout -> A-layout), 8 PV WMMAs.
__global__ __launch_bounds__(256) void attn_apply_kernel(const __bf16* __restrict__ Q,
                                                         const __bf16* __restrict__ Kmat,
                                                         const __bf16* __restrict__ Vt,
                                                         const float* __restrict__ Mcol,
                                                         const float* __restrict__ Lcol,
                                                         __bf16* __restrict__ Ctx) {
    __shared__ __bf16 Plds[8][32][40];   // per-wave 32x32 P tile, padded stride 40
    int bh = blockIdx.y;
    int b = bh / HEADS, h = bh % HEADS;
    int lane = threadIdx.x & 31;
    int wave = threadIdx.x >> 5;
    int i0 = blockIdx.x * 256 + wave * 32;

    const __bf16* Qh = Q    + (size_t)bh * SEQ * DK;
    const __bf16* Kh = Kmat + (size_t)bh * SEQ * DK;
    const __bf16* Vh = Vt   + (size_t)bh * DK * SEQ;  // [DK][SEQ] transposed
    const float*  Mh = Mcol + (size_t)bh * SEQ;
    const float*  Lh = Lcol + (size_t)bh * SEQ;

    // Q rows for this wave, resident for the whole j-loop
    v16bf aq[2][2];
#pragma unroll
    for (int sub = 0; sub < 2; ++sub) {
        aq[sub][0] = load_a_bf16(Qh + (size_t)(i0 + 16 * sub) * DK + 0,  DK, lane);
        aq[sub][1] = load_a_bf16(Qh + (size_t)(i0 + 16 * sub) * DK + 32, DK, lane);
    }

    int nl = lane & 15, half = (lane >> 4) & 1;
    v8f acc[2][4] = {};

    for (int j = 0; j < SEQ; j += 32) {
        // scores + probabilities for the two 16-wide j sub-tiles
#pragma unroll
        for (int jj = 0; jj < 2; ++jj) {
            int jb = j + 16 * jj;
            v16bf bk0 = load_b_t(Kh + (size_t)jb * DK + 0,  DK, lane);
            v16bf bk1 = load_b_t(Kh + (size_t)jb * DK + 32, DK, lane);
            float mc = Mh[jb + nl];
            float rl = 1.0f / Lh[jb + nl];
#pragma unroll
            for (int sub = 0; sub < 2; ++sub) {
                v8f c = {};
                c = wmma_bf16(aq[sub][0], bk0, c);
                c = wmma_bf16(aq[sub][1], bk1, c);
#pragma unroll
                for (int r = 0; r < 8; ++r) {
                    float p = __expf(c[r] - mc) * rl;
                    Plds[wave][16 * sub + r + 8 * half][16 * jj + nl] = (__bf16)p;
                }
            }
        }
        __syncthreads();   // C-layout -> A-layout cross-lane exchange via LDS

        v16bf ap0 = load_a_bf16(&Plds[wave][0][0],  40, lane);
        v16bf ap1 = load_a_bf16(&Plds[wave][16][0], 40, lane);
#pragma unroll
        for (int t = 0; t < 4; ++t) {
            // B[k=j][n=d] = V[j+k][16t+n] = Vt[16t+n][j+k], contiguous in k
            v16bf bv = load_b_t(Vh + (size_t)(16 * t) * SEQ + j, SEQ, lane);
            acc[0][t] = wmma_bf16(ap0, bv, acc[0][t]);
            acc[1][t] = wmma_bf16(ap1, bv, acc[1][t]);
        }
        __syncthreads();   // protect LDS tile before next iteration's writes
    }

    // Ctx layout [B,S,H,DK] (== flat [M, DMODEL]) so the output GEMM reads row-major
#pragma unroll
    for (int sub = 0; sub < 2; ++sub) {
#pragma unroll
        for (int t = 0; t < 4; ++t) {
#pragma unroll
            for (int r = 0; r < 8; ++r) {
                int i = i0 + 16 * sub + r + 8 * half;
                int d = 16 * t + nl;
                Ctx[((size_t)(b * SEQ + i) * HEADS + h) * DK + d] = (__bf16)acc[sub][t][r];
            }
        }
    }
}

// ---------------- host-side launcher ----------------
extern "C" void kernel_launch(void* const* d_in, const int* in_sizes, int n_in,
                              void* d_out, int out_size, void* d_ws, size_t ws_size,
                              hipStream_t stream) {
    (void)in_sizes; (void)n_in; (void)out_size; (void)ws_size;
    const float* query = (const float*)d_in[0];
    const float* key   = (const float*)d_in[1];
    const float* value = (const float*)d_in[2];
    const float* Wq = (const float*)d_in[3];
    const float* bq = (const float*)d_in[4];
    const float* Wk = (const float*)d_in[5];
    const float* bk = (const float*)d_in[6];
    const float* Wv = (const float*)d_in[7];
    const float* bv = (const float*)d_in[8];
    const float* Wo = (const float*)d_in[9];
    const float* bo = (const float*)d_in[10];

    char* ws = (char*)d_ws;
    size_t off = 0;
    const size_t W_ELEMS = (size_t)DMODEL * DMODEL;            // 1M
    const size_t X_ELEMS = (size_t)MROWS * DMODEL;             // 8M
    __bf16* Wqt = (__bf16*)(ws + off); off += W_ELEMS * 2;
    __bf16* Wkt = (__bf16*)(ws + off); off += W_ELEMS * 2;
    __bf16* Wvt = (__bf16*)(ws + off); off += W_ELEMS * 2;
    __bf16* Wot = (__bf16*)(ws + off); off += W_ELEMS * 2;
    __bf16* Qb  = (__bf16*)(ws + off); off += X_ELEMS * 2;     // [B,H,S,DK], pre-scaled
    __bf16* Kb  = (__bf16*)(ws + off); off += X_ELEMS * 2;     // [B,H,S,DK]
    __bf16* Vtb = (__bf16*)(ws + off); off += X_ELEMS * 2;     // [B,H,DK,S]
    __bf16* Ctx = (__bf16*)(ws + off); off += X_ELEMS * 2;     // [B,S,H,DK]
    float*  Mcol = (float*)(ws + off); off += (size_t)BATCH * HEADS * SEQ * 4;
    float*  Lcol = (float*)(ws + off); off += (size_t)BATCH * HEADS * SEQ * 4;

    dim3 blk(256, 1, 1);

    // 1) transpose weights (f32 -> bf16, [N][K])
    dim3 tgrid(DMODEL / 32, DMODEL / 32);
    transpose_w_kernel<<<tgrid, blk, 0, stream>>>(Wq, Wqt);
    transpose_w_kernel<<<tgrid, blk, 0, stream>>>(Wk, Wkt);
    transpose_w_kernel<<<tgrid, blk, 0, stream>>>(Wv, Wvt);
    transpose_w_kernel<<<tgrid, blk, 0, stream>>>(Wo, Wot);

    // 2) projections (Q carries the 1/sqrt(dk) scale; V stored transposed)
    dim3 ggrid(DMODEL / 128, MROWS / 128);
    gemm_kernel<0, true><<<ggrid, blk, 0, stream>>>(query, Wqt, bq, Qb, SCALE_Q);
    gemm_kernel<0, true><<<ggrid, blk, 0, stream>>>(key,   Wkt, bk, Kb, 1.0f);
    gemm_kernel<1, true><<<ggrid, blk, 0, stream>>>(value, Wvt, bv, Vtb, 1.0f);

    // 3) per-column (query-axis) softmax stats
    dim3 agrid(SEQ / 256, BATCH * HEADS);
    attn_stats_kernel<<<agrid, blk, 0, stream>>>(Qb, Kb, Mcol, Lcol);

    // 4) apply attention -> Ctx (bf16, [B,S,H*DK])
    attn_apply_kernel<<<agrid, blk, 0, stream>>>(Qb, Kb, Vtb, Mcol, Lcol, Ctx);

    // 5) output projection -> f32 d_out
    gemm_kernel<2, false><<<ggrid, blk, 0, stream>>>(Ctx, Wot, bo, (float*)d_out, 1.0f);
}
